// SelfAttention_62818191671602
// MI455X (gfx1250) — matrix operand order repaired and checked by reading, not compiled, not run on hardware
//
#include <hip/hip_runtime.h>
#include <hip/hip_bf16.h>
#include <math.h>

#define NB   8
#define CD   64
#define NPOS 4096           // 64*64
#define NTOT (NB*NPOS)
#define LDK  72             // LDS row stride (bf16 elems): 144B -> 16B aligned, conflict-free

typedef __bf16 bf16;
typedef __attribute__((ext_vector_type(8)))  float v8f;
typedef __attribute__((ext_vector_type(16))) bf16  v16bf;
typedef __attribute__((ext_vector_type(8)))  bf16  v8bf;

__device__ __forceinline__ bf16  f2bf(float f) { return (bf16)f; }
__device__ __forceinline__ float bf2f(bf16 h)  { return (float)h; }

// ---- DPP16 cross-lane helpers (VALU only; no LDS round trips) ------------
// ctrl: 0xB1 = quad_perm(1,0,3,2) -> xor1 ; 0x4E = quad_perm(2,3,0,1) -> xor2
//       0x141 = row_half_mirror (quad<->quad) ; 0x140 = row_mirror (octet<->octet)
template <int CTRL>
__device__ __forceinline__ float dppf(float v) {
  int r = __builtin_amdgcn_update_dpp(0, __builtin_bit_cast(int, v), CTRL, 0xf, 0xf, true);
  return __builtin_bit_cast(float, r);
}
__device__ __forceinline__ float row16_max(float v) {  // max over 16-lane group
  v = fmaxf(v, dppf<0xB1>(v));
  v = fmaxf(v, dppf<0x4E>(v));
  v = fmaxf(v, dppf<0x141>(v));
  v = fmaxf(v, dppf<0x140>(v));
  return v;
}
__device__ __forceinline__ float row16_sum(float v) {  // sum over 16-lane group
  v += dppf<0xB1>(v);
  v += dppf<0x4E>(v);
  v += dppf<0x141>(v);
  v += dppf<0x140>(v);
  return v;
}
__device__ __forceinline__ float quad_sum(float v) {   // sum over 4-lane group
  v += dppf<0xB1>(v);
  v += dppf<0x4E>(v);
  return v;
}

// ---- CDNA5 WMMA fragment loaders (16x16x32 bf16) -------------------------
// A-matrix 16x32 (MxK): lane lo holds row M=lo; lanes<16: K = kbase+{0..7,16..23};
// lanes>=16: K = kbase+{8..15, 24..31}.  Two contiguous 16B LDS reads.
__device__ __forceinline__ v16bf ldsA(const bf16* base, int row0, int kbase,
                                      int lo, int hi) {
  const bf16* p = base + (row0 + lo) * LDK + kbase + hi * 8;
  v16bf a;
  ((v8bf*)&a)[0] = *(const v8bf*)(p);
  ((v8bf*)&a)[1] = *(const v8bf*)(p + 16);
  return a;
}
// B-matrix 32x16 (KxN): lane lo holds column N=lo; lanes<16: K = kbase+{0..7,16..23};
// lanes>=16: K = kbase+{8..15, 24..31}.  base is indexed [col][k] (K contiguous).
__device__ __forceinline__ v16bf ldsB(const bf16* base, int col0, int kbase,
                                      int lo, int hi) {
  const bf16* p = base + (col0 + lo) * LDK + kbase + hi * 8;
  v16bf b;
  ((v8bf*)&b)[0] = *(const v8bf*)(p);
  ((v8bf*)&b)[1] = *(const v8bf*)(p + 16);
  return b;
}

// ---- K1: LayerNorm over channels + three 1x1 convs (NCHW out) ------------
__global__ void __launch_bounds__(256)
k_ln_qkv(const float* __restrict__ x, const float* __restrict__ lnw,
         const float* __restrict__ lnb, const float* __restrict__ wq,
         const float* __restrict__ wk, const float* __restrict__ wv,
         float* __restrict__ q1, float* __restrict__ k1, float* __restrict__ v1) {
  __shared__ float sWq[64 * 64], sWk[64 * 64], sWv[64 * 64];
  const int tid = threadIdx.x;
  for (int i = tid; i < 4096; i += 256) { sWq[i] = wq[i]; sWk[i] = wk[i]; sWv[i] = wv[i]; }
  __syncthreads();
  const int g = blockIdx.x * 256 + tid;        // 0..32767
  const int b = g >> 12, pos = g & 4095;
  const float* xb = x + (size_t)b * CD * NPOS + pos;
  float xr[64];
  float mu = 0.f;
#pragma unroll
  for (int c = 0; c < 64; ++c) { xr[c] = xb[c * 4096]; mu += xr[c]; }
  mu *= (1.f / 64.f);
  float var = 0.f;
#pragma unroll
  for (int c = 0; c < 64; ++c) { float d = xr[c] - mu; var += d * d; }
  var *= (1.f / 64.f);
  const float rstd = rsqrtf(var + 1e-5f);
#pragma unroll
  for (int c = 0; c < 64; ++c) xr[c] = (xr[c] - mu) * rstd * lnw[c] + lnb[c];
  const size_t ob = (size_t)b * CD * NPOS + pos;
  for (int co = 0; co < 64; ++co) {
    float aq = 0.f, ak = 0.f, av = 0.f;
#pragma unroll
    for (int c = 0; c < 64; ++c) {
      const float xv = xr[c];
      aq += sWq[co * 64 + c] * xv;
      ak += sWk[co * 64 + c] * xv;
      av += sWv[co * 64 + c] * xv;
    }
    q1[ob + co * 4096] = aq;
    k1[ob + co * 4096] = ak;
    v1[ob + co * 4096] = av;
  }
}

// ---- K2: depthwise 3x3, padding 1 (NCHW -> NCHW) -------------------------
__global__ void __launch_bounds__(256)
k_dwconv(const float* __restrict__ q1, const float* __restrict__ k1,
         const float* __restrict__ v1, const float* __restrict__ dq,
         const float* __restrict__ dk, const float* __restrict__ dv,
         float* __restrict__ qd, float* __restrict__ kd, float* __restrict__ vd) {
  const int e0 = blockIdx.x * 1024;
#pragma unroll
  for (int kk = 0; kk < 4; ++kk) {
    const int e = e0 + kk * 256 + threadIdx.x;         // < 8*64*4096
    const int b = e >> 18, c = (e >> 12) & 63, pos = e & 4095;
    const int h = pos >> 6, w = pos & 63;
    const size_t base = (size_t)b * 262144 + (size_t)c * 4096;
    float aq = 0.f, ak = 0.f, av = 0.f;
#pragma unroll
    for (int ky = 0; ky < 3; ++ky) {
      const int hy = h + ky - 1;
#pragma unroll
      for (int kx = 0; kx < 3; ++kx) {
        const int wx = w + kx - 1;
        if (hy >= 0 && hy < 64 && wx >= 0 && wx < 64) {
          const size_t gi = base + hy * 64 + wx;
          const int wi = c * 9 + ky * 3 + kx;
          aq += q1[gi] * dq[wi];
          ak += k1[gi] * dk[wi];
          av += v1[gi] * dv[wi];
        }
      }
    }
    qd[base + pos] = aq; kd[base + pos] = ak; vd[base + pos] = av;
  }
}

// ---- K3: tokenize. q/k: token-major [b][n][c] + l2norm. v: channel-major --
__global__ void __launch_bounds__(256)
k_tok(const float* __restrict__ qd, const float* __restrict__ kd,
      const float* __restrict__ vd, bf16* __restrict__ qb,
      bf16* __restrict__ kb, bf16* __restrict__ vb) {
  __shared__ float Tq[64][65], Tk[64][65], Tv[64][65];
  const int blk = blockIdx.x, b = blk >> 6, pos0 = (blk & 63) * 64;
  const int tid = threadIdx.x;
  for (int j = tid; j < 4096; j += 256) {
    const int c = j >> 6, p = j & 63;
    const size_t gi = (size_t)b * 262144 + (size_t)c * 4096 + pos0 + p;
    Tq[c][p] = qd[gi]; Tk[c][p] = kd[gi]; Tv[c][p] = vd[gi];
  }
  __syncthreads();
  {
    const int p = tid >> 2, qq = tid & 3;
    float sq = 0.f, sk = 0.f;
#pragma unroll
    for (int i = 0; i < 16; ++i) {
      const int c = qq * 16 + i;
      const float a = Tq[c][p]; sq += a * a;
      const float k = Tk[c][p]; sk += k * k;
    }
    sq = quad_sum(sq);
    sk = quad_sum(sk);
    const float iq = 1.f / fmaxf(sqrtf(sq), 1e-12f);
    const float ik = 1.f / fmaxf(sqrtf(sk), 1e-12f);
    const size_t ob = ((size_t)b * NPOS + pos0 + p) * 64 + qq * 16;
#pragma unroll
    for (int i = 0; i < 16; ++i) {
      const int c = qq * 16 + i;
      qb[ob + i] = f2bf(Tq[c][p] * iq);
      kb[ob + i] = f2bf(Tk[c][p] * ik);
    }
  }
  { // v: channel-major bf16 [b][c][4096] so flash stages it with b128 copies
    const int c = tid & 63, pg = tid >> 6;        // 4 groups of 16 positions
    const size_t vo = ((size_t)b * 64 + c) * (size_t)NPOS + pos0 + pg * 16;
#pragma unroll
    for (int i = 0; i < 16; ++i) vb[vo + i] = f2bf(Tv[c][pg * 16 + i]);
  }
}

// ---- K4: per-batch stats G=K^T K, H=Q^T Q, column sums (atomics) ---------
__global__ void __launch_bounds__(256)
k_stats(const bf16* __restrict__ qb, const bf16* __restrict__ kb,
        float* __restrict__ G, float* __restrict__ Hm,
        float* __restrict__ qs, float* __restrict__ ks) {
  __shared__ float Qf[64][65], Kf[64][65];
  const int blk = blockIdx.x, b = blk >> 6, pos0 = (blk & 63) * 64;
  const int tid = threadIdx.x;
  for (int j = tid; j < 4096; j += 256) {
    const int p = j >> 6, c = j & 63;
    const size_t gi = ((size_t)b * NPOS + pos0 + p) * 64 + c;
    Qf[p][c] = bf2f(qb[gi]); Kf[p][c] = bf2f(kb[gi]);
  }
  __syncthreads();
  for (int it = 0; it < 16; ++it) {
    const int idx = it * 256 + tid, c1 = idx >> 6, c2 = idx & 63;
    float g = 0.f, h = 0.f;
#pragma unroll 8
    for (int p = 0; p < 64; ++p) { g += Kf[p][c1] * Kf[p][c2]; h += Qf[p][c1] * Qf[p][c2]; }
    atomicAdd(&G[b * 4096 + idx], g);
    atomicAdd(&Hm[b * 4096 + idx], h);
  }
  if (tid < 64) {
    float s = 0.f;
    for (int p = 0; p < 64; ++p) s += Kf[p][tid];
    atomicAdd(&ks[b * 64 + tid], s);
  } else if (tid < 128) {
    const int c = tid - 64;
    float s = 0.f;
    for (int p = 0; p < 64; ++p) s += Qf[p][c];
    atomicAdd(&qs[b * 64 + c], s);
  }
}

// ---- K5: finalize per-batch logit scale = 1/(sqrt(n)*sigma) --------------
__global__ void __launch_bounds__(256)
k_scale(const float* __restrict__ G, const float* __restrict__ Hm,
        const float* __restrict__ qs, const float* __restrict__ ks,
        float* __restrict__ scale) {
  __shared__ float red[256];
  const int b = blockIdx.x, tid = threadIdx.x;
  float acc = 0.f;
  for (int i = tid; i < 4096; i += 256) acc += G[b * 4096 + i] * Hm[b * 4096 + i];
  red[tid] = acc; __syncthreads();
  for (int s = 128; s > 0; s >>= 1) { if (tid < s) red[tid] += red[tid + s]; __syncthreads(); }
  const float sumR2 = red[0];
  __syncthreads();
  red[tid] = (tid < 64) ? qs[b * 64 + tid] * ks[b * 64 + tid] : 0.f;
  __syncthreads();
  for (int s = 128; s > 0; s >>= 1) { if (tid < s) red[tid] += red[tid + s]; __syncthreads(); }
  if (tid == 0) {
    const float sumR = red[0];
    const float n = 4096.f, sn = sqrtf(n);
    const float mu  = sumR / (n * n * sn);            // mean of S = R/sqrt(n)
    const float es2 = sumR2 / (n * n * n);            // E[S^2]
    const float var = fmaxf(es2 - mu * mu, 0.f);
    const float sigma = sqrtf(var + 1e-5f);
    scale[b] = 1.f / (sn * sigma);                    // softmax shift-invariance kills mu
  }
}

// ---- K6: flash attention, bf16 WMMA, f32 accumulate ----------------------
// 256 threads = 8 waves, 128 query rows/block, 64-wide KV tiles,
// software-pipelined global->reg->LDS staging + prefetch of tile t+2.
__global__ void __launch_bounds__(256)
k_flash(const bf16* __restrict__ qb, const bf16* __restrict__ kb,
        const bf16* __restrict__ vb, const float* __restrict__ scale_g,
        float* __restrict__ attn) {
  __shared__ __align__(16) bf16 sQ[128 * LDK];         // [token][c]
  __shared__ __align__(16) bf16 sK[64 * LDK];          // [token][c]
  __shared__ __align__(16) bf16 sVT[64 * LDK];         // [c][token]
  __shared__ __align__(16) bf16 sP[8 * 16 * LDK];      // per-wave P tiles

  const int tid = threadIdx.x;
  const int wave = tid >> 5, lane = tid & 31;
  const int lo = lane & 15, hi = lane >> 4;
  const int b = blockIdx.y;
  const int q0 = blockIdx.x * 128;
  const float scale = scale_g[b];

  { // stage Q tile (128 rows x 64 ch)
    const int row = tid >> 1, half = tid & 1;
    const bf16* src = qb + ((size_t)(b * NPOS + q0 + row)) * 64 + half * 32;
    bf16* dst = sQ + row * LDK + half * 32;
#pragma unroll
    for (int i = 0; i < 4; ++i) *(v8bf*)(dst + i * 8) = *(const v8bf*)(src + i * 8);
  }
  __syncthreads();
  const v16bf qa0 = ldsA(sQ, wave * 16, 0, lo, hi);
  const v16bf qa1 = ldsA(sQ, wave * 16, 32, lo, hi);

  v8f o[4];
  float rmax[8], rsum[8];
#pragma unroll
  for (int ct = 0; ct < 4; ++ct)
#pragma unroll
    for (int r = 0; r < 8; ++r) o[ct][r] = 0.f;
#pragma unroll
  for (int r = 0; r < 8; ++r) { rmax[r] = -1e30f; rsum[r] = 0.f; }

  // tile staging: 256 threads cover 64x64; srow = token row (K) / channel row (VT)
  const int srow = tid >> 2, sq4 = tid & 3;            // 16 elems per thread
  const bf16* kbase = kb + ((size_t)b * NPOS) * 64;
  const bf16* vbase = vb + ((size_t)b * 64) * (size_t)NPOS;
  v8bf kreg[2], vreg[2];
  auto ldtile = [&](int t) {
    const bf16* kp = kbase + ((size_t)(t * 64 + srow)) * 64 + sq4 * 16;
    const bf16* vp = vbase + (size_t)srow * NPOS + t * 64 + sq4 * 16;
    kreg[0] = ((const v8bf*)kp)[0]; kreg[1] = ((const v8bf*)kp)[1];
    vreg[0] = ((const v8bf*)vp)[0]; vreg[1] = ((const v8bf*)vp)[1];
  };
  ldtile(0);

  const int NT = NPOS / 64;
  for (int t = 0; t < NT; ++t) {
    { // regs -> LDS
      bf16* kd = sK  + srow * LDK + sq4 * 16;
      bf16* vd = sVT + srow * LDK + sq4 * 16;
      ((v8bf*)kd)[0] = kreg[0]; ((v8bf*)kd)[1] = kreg[1];
      ((v8bf*)vd)[0] = vreg[0]; ((v8bf*)vd)[1] = vreg[1];
    }
    __syncthreads();
    if (t + 1 < NT) ldtile(t + 1);                     // overlap with compute
    if (t + 2 < NT) {                                  // gfx1250 global_prefetch
      __builtin_prefetch(kbase + ((size_t)((t + 2) * 64 + srow)) * 64 + sq4 * 16, 0, 1);
      __builtin_prefetch(vbase + (size_t)srow * NPOS + (t + 2) * 64 + sq4 * 16, 0, 1);
    }

    // S = scale * Q K^T  (16 rows x 64 cols per wave)
    v8f s[4];
#pragma unroll
    for (int nt = 0; nt < 4; ++nt) {
      const v16bf bk0 = ldsB(sK, nt * 16, 0, lo, hi);
      const v16bf bk1 = ldsB(sK, nt * 16, 32, lo, hi);
      v8f acc;
#pragma unroll
      for (int r = 0; r < 8; ++r) acc[r] = 0.f;
      acc = __builtin_amdgcn_wmma_f32_16x16x32_bf16(false, qa0, false, bk0, (short)0, acc, false, false);
      acc = __builtin_amdgcn_wmma_f32_16x16x32_bf16(false, qa1, false, bk1, (short)0, acc, false, false);
#pragma unroll
      for (int r = 0; r < 8; ++r) s[nt][r] = acc[r] * scale;
    }

    // online softmax: row reductions via DPP (no LDS)
    float ls[8];
#pragma unroll
    for (int r = 0; r < 8; ++r) {
      float tm = fmaxf(fmaxf(s[0][r], s[1][r]), fmaxf(s[2][r], s[3][r]));
      tm = row16_max(tm);
      const float nm = fmaxf(rmax[r], tm);
      const float cr = __expf(rmax[r] - nm);
      rmax[r] = nm;
      rsum[r] *= cr;
      ls[r] = 0.f;
#pragma unroll
      for (int ct = 0; ct < 4; ++ct) o[ct][r] *= cr;
    }
#pragma unroll
    for (int nt = 0; nt < 4; ++nt)
#pragma unroll
      for (int r = 0; r < 8; ++r) {
        const float p = __expf(s[nt][r] - rmax[r]);
        s[nt][r] = p;
        ls[r] += p;
      }
#pragma unroll
    for (int r = 0; r < 8; ++r) rsum[r] += row16_sum(ls[r]);

    // P -> LDS (per-wave region; wave-local RAW, DS in-order) then PV
    bf16* myP = sP + wave * 16 * LDK;
#pragma unroll
    for (int nt = 0; nt < 4; ++nt)
#pragma unroll
      for (int r = 0; r < 8; ++r)
        myP[(r + hi * 8) * LDK + nt * 16 + lo] = f2bf(s[nt][r]);
    const v16bf pa0 = ldsA(myP, 0, 0, lo, hi);
    const v16bf pa1 = ldsA(myP, 0, 32, lo, hi);
#pragma unroll
    for (int ct = 0; ct < 4; ++ct) {
      const v16bf bv0 = ldsB(sVT, ct * 16, 0, lo, hi);
      const v16bf bv1 = ldsB(sVT, ct * 16, 32, lo, hi);
      o[ct] = __builtin_amdgcn_wmma_f32_16x16x32_bf16(false, pa0, false, bv0, (short)0, o[ct], false, false);
      o[ct] = __builtin_amdgcn_wmma_f32_16x16x32_bf16(false, pa1, false, bv1, (short)0, o[ct], false, false);
    }
    __syncthreads();                                   // protect K/V LDS for next tile
  }

  // epilogue: normalize rows, write token-major f32
#pragma unroll
  for (int ct = 0; ct < 4; ++ct)
#pragma unroll
    for (int r = 0; r < 8; ++r) {
      const int row = q0 + wave * 16 + hi * 8 + r;
      attn[((size_t)b * NPOS + row) * 64 + ct * 16 + lo] = o[ct][r] / rsum[r];
    }
}

// ---- K7: 1x1 proj + bias + residual (token-major -> NCHW) ----------------
__global__ void __launch_bounds__(256)
k_proj(const float* __restrict__ attn, const float* __restrict__ pw,
       const float* __restrict__ pb, const float* __restrict__ x,
       float* __restrict__ out) {
  __shared__ float At[64][65];
  __shared__ float Wt[64][65];
  const int blk = blockIdx.x, b = blk >> 6, pos0 = (blk & 63) * 64;
  const int tid = threadIdx.x;
  for (int j = tid; j < 4096; j += 256) {
    const int p = j >> 6, c = j & 63;
    At[p][c] = attn[((size_t)b * NPOS + pos0 + p) * 64 + c];
    Wt[p][c] = pw[j];
  }
  __syncthreads();
  const int p = tid & 63, cog = tid >> 6;
  for (int it = 0; it < 16; ++it) {
    const int co = it * 4 + cog;
    float acc = pb[co];
#pragma unroll 8
    for (int c = 0; c < 64; ++c) acc += Wt[co][c] * At[p][c];
    const size_t oi = (size_t)b * 262144 + (size_t)co * 4096 + pos0 + p;
    out[oi] = x[oi] + acc;
  }
}

// ---- launch --------------------------------------------------------------
extern "C" void kernel_launch(void* const* d_in, const int* in_sizes, int n_in,
                              void* d_out, int out_size, void* d_ws, size_t ws_size,
                              hipStream_t stream) {
  const float* x   = (const float*)d_in[0];
  const float* lnw = (const float*)d_in[1];
  const float* lnb = (const float*)d_in[2];
  const float* wq  = (const float*)d_in[3];
  const float* wk  = (const float*)d_in[4];
  const float* wv  = (const float*)d_in[5];
  const float* dq  = (const float*)d_in[6];
  const float* dk  = (const float*)d_in[7];
  const float* dv  = (const float*)d_in[8];
  const float* pw  = (const float*)d_in[9];
  const float* pb  = (const float*)d_in[10];

  char* ws = (char*)d_ws;
  size_t off = 0;
  auto alloc = [&](size_t bytes) { size_t o = off; off = (off + bytes + 255) & ~(size_t)255; return o; };
  const size_t NCHW = (size_t)NB * CD * NPOS * 4;       // 8.4 MB
  const size_t TOKB = (size_t)NB * NPOS * CD * 2;       // bf16
  const size_t o_q1 = alloc(NCHW), o_k1 = alloc(NCHW), o_v1 = alloc(NCHW);
  const size_t o_qd = alloc(NCHW), o_kd = alloc(NCHW), o_vd = alloc(NCHW);
  const size_t o_qb = alloc(TOKB), o_kb = alloc(TOKB), o_vb = alloc(TOKB);
  const size_t o_G  = alloc((size_t)NB * 4096 * 4);
  const size_t o_H  = alloc((size_t)NB * 4096 * 4);
  const size_t o_qs = alloc((size_t)NB * 64 * 4);
  const size_t o_ks = alloc((size_t)NB * 64 * 4);
  const size_t o_sc = alloc((size_t)NB * 4);
  const size_t o_at = alloc(NCHW);

  float* q1 = (float*)(ws + o_q1); float* k1 = (float*)(ws + o_k1); float* v1 = (float*)(ws + o_v1);
  float* qd = (float*)(ws + o_qd); float* kd = (float*)(ws + o_kd); float* vd = (float*)(ws + o_vd);
  bf16*  qb = (bf16*)(ws + o_qb);  bf16*  kb = (bf16*)(ws + o_kb);  bf16*  vb = (bf16*)(ws + o_vb);
  float* G  = (float*)(ws + o_G);  float* Hm = (float*)(ws + o_H);
  float* qs = (float*)(ws + o_qs); float* ks = (float*)(ws + o_ks);
  float* sc = (float*)(ws + o_sc); float* at = (float*)(ws + o_at);

  hipMemsetAsync(ws + o_G, 0, (o_sc - o_G), stream);    // zero G,H,qsum,ksum

  k_ln_qkv<<<NTOT / 256, 256, 0, stream>>>(x, lnw, lnb, wq, wk, wv, q1, k1, v1);
  k_dwconv<<<(NB * CD * NPOS) / 1024, 256, 0, stream>>>(q1, k1, v1, dq, dk, dv, qd, kd, vd);
  k_tok<<<NTOT / 64, 256, 0, stream>>>(qd, kd, vd, qb, kb, vb);
  k_stats<<<NTOT / 64, 256, 0, stream>>>(qb, kb, G, Hm, qs, ks);
  k_scale<<<NB, 256, 0, stream>>>(G, Hm, qs, ks, sc);
  k_flash<<<dim3(NPOS / 128, NB), 256, 0, stream>>>(qb, kb, vb, sc, at);
  k_proj<<<NTOT / 64, 256, 0, stream>>>(at, pw, pb, x, (float*)d_out);
}
